// tcnnshader_82669530513591
// MI455X (gfx1250) — matrix operand order repaired and checked by reading, not compiled, not run on hardware
//
#include <hip/hip_runtime.h>
#include <stdint.h>

// ---------------------------------------------------------------------------
// CDNA5 wave32 WMMA f16 helpers (16x16x32, f32 accumulate)
// ---------------------------------------------------------------------------
typedef __attribute__((ext_vector_type(16))) _Float16 v16h;
typedef __attribute__((ext_vector_type(8)))  float    v8f;

union AFrag { v16h v; uint32_t u[8]; };

__device__ __forceinline__ v8f wmma_f16(v16h a, v16h b, v8f c) {
  // 8 args: (neg_a, A, neg_b, B, c_mod, C, reuse_a, reuse_b)
  return __builtin_amdgcn_wmma_f32_16x16x32_f16(false, a, false, b, (short)0, c,
                                                false, false);
}

// A-matrix 16x32 f16 fragment from a row-major half buffer (LDS or global).
// ISA layout: lanes 0-15 rows M=0..15, VGPR v: K = (v<4 ? 2v : 2v+8) + 8*(lane>=16)
// => per-lane dwords {0,1,2,3,8,9,10,11} + 4*(lane>=16) of the 32-half k-slab.
__device__ __forceinline__ v16h load_a_frag(const _Float16* rowBase, int strideH,
                                            int ktile) {
  const int lane = threadIdx.x & 31;
  const int r = lane & 15, hi = lane >> 4;
  const uint32_t* p =
      (const uint32_t*)(rowBase + (size_t)r * strideH + ktile * 32) + 4 * hi;
  AFrag f;
  f.u[0] = p[0];  f.u[1] = p[1];  f.u[2] = p[2];  f.u[3] = p[3];
  f.u[4] = p[8];  f.u[5] = p[9];  f.u[6] = p[10]; f.u[7] = p[11];
  return f.v;
}

// B-matrix 32x16 fragment from pre-tiled weights: 1024B block, lane-major
// (lane L owns 16 contiguous f16 = its 8 VGPRs) -> single 32B load per lane.
__device__ __forceinline__ v16h load_b_frag(const _Float16* block) {
  const int lane = threadIdx.x & 31;
  return *(const v16h*)(block + lane * 16);
}

// C/D f32 tile: lane holds col N = lane&15; VGPR g holds row M = g + 8*(lane>=16).
__device__ __forceinline__ void store_tile(_Float16* outBase, int strideH,
                                           int rowBase, int nBase, v8f c,
                                           const float* bias, bool relu) {
  const int lane = threadIdx.x & 31;
  const int n = lane & 15, hi = lane >> 4;
  const float b = bias[nBase + n];
  #pragma unroll
  for (int g = 0; g < 8; ++g) {
    float v = c[g] + b;
    if (relu) v = fmaxf(v, 0.f);
    outBase[(size_t)(rowBase + g + 8 * hi) * strideH + nBase + n] = (_Float16)v;
  }
}

// CDNA5 async DMA: global -> LDS, 16B per lane, tracked by ASYNCcnt.
__device__ __forceinline__ void async_copy_b128(uint32_t ldsByteAddr,
                                                const void* gaddr) {
  asm volatile("global_load_async_to_lds_b128 %0, %1, off"
               :: "v"(ldsByteAddr), "v"(gaddr)
               : "memory");
}
__device__ __forceinline__ void wait_asynccnt0() {
  asm volatile("s_wait_asynccnt 0x0" ::: "memory");
}

// ---------------------------------------------------------------------------
// Weight pre-tiling: f32 [K][N] row-major -> f16 blocks of 512 halves laid out
// in exact B-fragment lane order. Block index = kt*ntiles + nt.
// Element (lane, idx): k = kt*32 + 8*(lane>=16) + idx + 8*(idx>=8), n = nt*16 + (lane&15).
// ---------------------------------------------------------------------------
__global__ __launch_bounds__(512) void tile_weights_kernel(
    const float* __restrict__ W, int K, int Nd, _Float16* __restrict__ dst,
    int ntiles) {
  const int bt = blockIdx.x;
  const int kt = bt / ntiles, nt = bt % ntiles;
  const int t = threadIdx.x;
  const int lane = t >> 4, idx = t & 15;
  const int k = kt * 32 + 8 * (lane >> 4) + idx + 8 * (idx >> 3);
  const int n = nt * 16 + (lane & 15);
  const float v = (k < K && n < Nd) ? W[(size_t)k * Nd + n] : 0.f;
  dst[(size_t)bt * 512 + lane * 16 + idx] = (_Float16)v;
}

// ---------------------------------------------------------------------------
// Encode: ndir/wr/mask, freq encoding (padded to 64), hash-grid encoding.
// ---------------------------------------------------------------------------
#define TABLE_SIZE 524288
#define L_LEVELS 16

__global__ __launch_bounds__(256) void encode_kernel(
    const float* __restrict__ x, const float* __restrict__ nrm,
    const float* __restrict__ dv, const float* __restrict__ grid,
    _Float16* __restrict__ pe, _Float16* __restrict__ feat,
    float* __restrict__ ndirOut, int* __restrict__ maskOut, int N) {
  const int i = blockIdx.x * 256 + threadIdx.x;
  if (i >= N) return;
  const float x0 = x[3 * i + 0], x1 = x[3 * i + 1], x2 = x[3 * i + 2];
  const float n0 = nrm[3 * i + 0], n1 = nrm[3 * i + 1], n2 = nrm[3 * i + 2];
  const float d0 = dv[3 * i + 0], d1 = dv[3 * i + 1], d2 = dv[3 * i + 2];

  const float nd = d0 * n0 + d1 * n1 + d2 * n2;
  const float w[3] = {2.f * nd * n0 - d0, 2.f * nd * n1 - d1, 2.f * nd * n2 - d2};
  ndirOut[i] = nd;
  maskOut[i] = (x0 > -0.5f && x0 < 0.5f && x1 > -0.5f && x1 < 0.5f &&
                x2 > -0.5f && x2 < 0.5f) ? 1 : 0;

  // frequency encoding: pe[c*12 + f] = sin, pe[c*12 + 6 + f] = cos
  _Float16* p = pe + (size_t)i * 64;
  #pragma unroll
  for (int c = 0; c < 3; ++c) {
    float fr = 3.14159265358979f;
    #pragma unroll
    for (int f = 0; f < 6; ++f) {
      const float ang = w[c] * fr;
      p[c * 12 + f] = (_Float16)__sinf(ang);
      p[c * 12 + 6 + f] = (_Float16)__cosf(ang);
      fr *= 2.f;
    }
  }
  #pragma unroll
  for (int j = 36; j < 64; ++j) p[j] = (_Float16)0.f;

  // hash-grid encoding
  const float g0 = (x0 + 1.f) * 0.5f;
  const float g1 = (x1 + 1.f) * 0.5f;
  const float g2 = (x2 + 1.f) * 0.5f;
  const float bgrow = __expf(__logf(32.f) / 15.f);  // (512/16)^(1/15)
  float resf = 16.f;
  _Float16* fo = feat + (size_t)i * 32;
  for (int l = 0; l < L_LEVELS; ++l) {
    const float r = floorf(resf);
    const float px = g0 * r, py = g1 * r, pz = g2 * r;
    const float fx0 = floorf(px), fy0 = floorf(py), fz0 = floorf(pz);
    const float fx = px - fx0, fy = py - fy0, fz = pz - fz0;
    const int ix = (int)fx0, iy = (int)fy0, iz = (int)fz0;
    float a0 = 0.f, a1 = 0.f;
    #pragma unroll
    for (int c = 0; c < 8; ++c) {
      const int cx = (c >> 2) & 1, cy = (c >> 1) & 1, cz = c & 1;
      const uint32_t hx = (uint32_t)(ix + cx);
      const uint32_t hy = (uint32_t)(iy + cy) * 2654435761u;
      const uint32_t hz = (uint32_t)(iz + cz) * 805459861u;
      const uint32_t idx = (hx ^ hy ^ hz) & (uint32_t)(TABLE_SIZE - 1);
      const float* g = grid + ((size_t)l * TABLE_SIZE + idx) * 2;
      const float wt = (cx ? fx : 1.f - fx) * (cy ? fy : 1.f - fy) *
                       (cz ? fz : 1.f - fz);
      a0 += wt * g[0];
      a1 += wt * g[1];
    }
    fo[l * 2 + 0] = (_Float16)a0;
    fo[l * 2 + 1] = (_Float16)a1;
    resf *= bgrow;
  }
}

// ---------------------------------------------------------------------------
// Hash MLP: 32 -> 64 (relu) -> 6 (sigmoid). 128 rows/block, wave owns 16 rows
// (activations are wave-private -> no barriers needed).
// ---------------------------------------------------------------------------
__global__ __launch_bounds__(256) void hash_mlp_kernel(
    const _Float16* __restrict__ feat, const _Float16* __restrict__ wt,
    const float* __restrict__ b1, const float* __restrict__ b2,
    float* __restrict__ hOut, float* __restrict__ dOut, int N) {
  __shared__ __align__(16) _Float16 act[128 * 72];
  const int tid = threadIdx.x, wave = tid >> 5, lane = tid & 31;
  const int R0 = blockIdx.x * 128 + wave * 16;

  {  // L1: K=32 (1 k-tile), N=64 (4 n-tiles)
    v8f a0 = {}, a1 = {}, a2 = {}, a3 = {};
    v16h a = load_a_frag(feat + (size_t)R0 * 32, 32, 0);
    a0 = wmma_f16(a, load_b_frag(wt + 0 * 512), a0);
    a1 = wmma_f16(a, load_b_frag(wt + 1 * 512), a1);
    a2 = wmma_f16(a, load_b_frag(wt + 2 * 512), a2);
    a3 = wmma_f16(a, load_b_frag(wt + 3 * 512), a3);
    store_tile(act, 72, wave * 16, 0, a0, b1, true);
    store_tile(act, 72, wave * 16, 16, a1, b1, true);
    store_tile(act, 72, wave * 16, 32, a2, b1, true);
    store_tile(act, 72, wave * 16, 48, a3, b1, true);
  }
  {  // L2: K=64 (2 k-tiles), one n-tile, 6 valid cols
    const _Float16* w2 = wt + 4 * 512;
    v8f acc = {};
    acc = wmma_f16(load_a_frag(act + wave * 16 * 72, 72, 0), load_b_frag(w2), acc);
    acc = wmma_f16(load_a_frag(act + wave * 16 * 72, 72, 1), load_b_frag(w2 + 512), acc);
    const int n = lane & 15, hi = lane >> 4;
    if (n < 6) {
      const float b = b2[n];
      #pragma unroll
      for (int g = 0; g < 8; ++g) {
        const int row = R0 + g + 8 * hi;
        const float v = 1.f / (1.f + __expf(-(acc[g] + b)));
        hOut[(size_t)row * 6 + n] = v;
        if (n < 3) dOut[1 + (size_t)6 * N + (size_t)row * 3 + n] = v;  // diffuse
      }
    }
  }
}

// ---------------------------------------------------------------------------
// Env MLP: 36(pad 64) -> 256 -> 256 -> 256 -> 256 -> 3, relu hidden.
// 128 rows/block, 8 waves; wave w owns rows 16w..16w+15 and computes ALL 16
// n-tiles (16 live accumulator tiles = 128 VGPRs) -> 4x less weight traffic
// than 32-row tiles, and all LDS activity is wave-private (zero barriers).
// Input staged via CDNA5 async DMA (global_load_async_to_lds_b128).
// Launched twice (branch 1 = env_params on mask, 0 = env1 on !mask).
// ---------------------------------------------------------------------------
struct Bias5 { const float *b1, *b2, *b3, *b4, *b5; };

__device__ __forceinline__ void env_layer128(const _Float16* aIn, _Float16* aOut,
                                             const _Float16* wt, const float* bias,
                                             int ktiles, int rbase) {
  v8f acc[16] = {};
  for (int kt = 0; kt < ktiles; ++kt) {
    v16h a = load_a_frag(aIn + (size_t)rbase * 264, 264, kt);
    const _Float16* wk = wt + (size_t)kt * 16 * 512;
    __builtin_prefetch(wk + 16 * 512, 0, 0);  // next k-slab -> global_prefetch
    #pragma unroll
    for (int nt = 0; nt < 16; ++nt)
      acc[nt] = wmma_f16(a, load_b_frag(wk + nt * 512), acc[nt]);
  }
  #pragma unroll
  for (int nt = 0; nt < 16; ++nt)
    store_tile(aOut, 264, rbase, nt * 16, acc[nt], bias, true);
}

__global__ __launch_bounds__(256) void env_mlp_kernel(
    const _Float16* __restrict__ pe, const _Float16* __restrict__ wt, Bias5 bs,
    const int* __restrict__ mask, float* __restrict__ envOut, int wantMask) {
  __shared__ __align__(16) _Float16 act[2][128 * 264];
  const int tid = threadIdx.x, wave = tid >> 5, lane = tid & 31;
  const int rbase = wave * 16;
  const int R0 = blockIdx.x * 128;

  // Async-stage this wave's 16 rows of padded PE input (16 x 128B).
  // 128 chunks of 16B per wave -> 4 per lane. Wave-private rows, so only an
  // ASYNCcnt wait is needed (no workgroup barrier).
  #pragma unroll
  for (int i = 0; i < 4; ++i) {
    const int chunk = lane + 32 * i;          // 0..127
    const int r = rbase + (chunk >> 3);       // this wave's rows
    const int c = chunk & 7;                  // 16B chunk within 128B row
    const uint32_t lds =
        (uint32_t)(uintptr_t)(&act[0][(size_t)r * 264]) + c * 16;
    async_copy_b128(lds, pe + (size_t)(R0 + r) * 64 + c * 8);
  }
  wait_asynccnt0();

  const _Float16* w1 = wt;              // 2 kt x 16 nt
  const _Float16* w2 = wt + 32 * 512;   // 8 x 16
  const _Float16* w3 = wt + 160 * 512;  // 8 x 16
  const _Float16* w4 = wt + 288 * 512;  // 8 x 16
  const _Float16* w5 = wt + 416 * 512;  // 8 x 1

  env_layer128(act[0], act[1], w1, bs.b1, 2, rbase);
  env_layer128(act[1], act[0], w2, bs.b2, 8, rbase);
  env_layer128(act[0], act[1], w3, bs.b3, 8, rbase);
  env_layer128(act[1], act[0], w4, bs.b4, 8, rbase);

  // Final 256 -> 3: every wave handles its own rows, single n-tile.
  v8f acc = {};
  #pragma unroll
  for (int kt = 0; kt < 8; ++kt) {
    v16h a = load_a_frag(&act[0][(size_t)rbase * 264], 264, kt);
    acc = wmma_f16(a, load_b_frag(w5 + kt * 512), acc);
  }
  const int n = lane & 15, hi = lane >> 4;
  if (n < 3) {
    const float b = bs.b5[n];
    #pragma unroll
    for (int g = 0; g < 8; ++g) {
      const int row = R0 + rbase + g + 8 * hi;
      if ((mask[row] != 0) == (wantMask != 0))
        envOut[(size_t)row * 3 + n] = acc[g] + b;
    }
  }
}

// ---------------------------------------------------------------------------
// RGB MLP: 7(pad 32) -> 64 -> 64 -> 3, sigmoid; then clips + partial sums.
// ---------------------------------------------------------------------------
__global__ __launch_bounds__(256) void rgb_mlp_kernel(
    const float* __restrict__ env, const float* __restrict__ hFeat,
    const float* __restrict__ ndir, const _Float16* __restrict__ wt,
    const float* __restrict__ b1, const float* __restrict__ b2,
    const float* __restrict__ b3, float* __restrict__ dOut,
    float* __restrict__ parts, int N) {
  __shared__ __align__(16) _Float16 inA[128 * 32];
  __shared__ __align__(16) _Float16 t1[128 * 72];
  __shared__ __align__(16) _Float16 t2[128 * 72];
  __shared__ float red[256];
  const int tid = threadIdx.x, wave = tid >> 5, lane = tid & 31;
  const int R0 = blockIdx.x * 128;

  if (tid < 128) {
    const int row = R0 + tid;
    _Float16* p = inA + tid * 32;
    p[0] = (_Float16)env[(size_t)row * 3 + 0];
    p[1] = (_Float16)env[(size_t)row * 3 + 1];
    p[2] = (_Float16)env[(size_t)row * 3 + 2];
    p[3] = (_Float16)hFeat[(size_t)row * 6 + 3];
    p[4] = (_Float16)hFeat[(size_t)row * 6 + 4];
    p[5] = (_Float16)hFeat[(size_t)row * 6 + 5];
    p[6] = (_Float16)ndir[row];
    #pragma unroll
    for (int j = 7; j < 32; ++j) p[j] = (_Float16)0.f;
  }
  __syncthreads();  // staging is cross-wave

  const int rbase = wave * 16;
  {  // L1: K=32(1 kt), 4 n-tiles
    v8f a0 = {}, a1 = {}, a2 = {}, a3 = {};
    v16h a = load_a_frag(inA + rbase * 32, 32, 0);
    a0 = wmma_f16(a, load_b_frag(wt + 0 * 512), a0);
    a1 = wmma_f16(a, load_b_frag(wt + 1 * 512), a1);
    a2 = wmma_f16(a, load_b_frag(wt + 2 * 512), a2);
    a3 = wmma_f16(a, load_b_frag(wt + 3 * 512), a3);
    store_tile(t1, 72, rbase, 0, a0, b1, true);
    store_tile(t1, 72, rbase, 16, a1, b1, true);
    store_tile(t1, 72, rbase, 32, a2, b1, true);
    store_tile(t1, 72, rbase, 48, a3, b1, true);
  }
  {  // L2: K=64(2 kt), 4 n-tiles; blocks at wt + (4 + kt*4 + nt)*512
    v8f a0 = {}, a1 = {}, a2 = {}, a3 = {};
    #pragma unroll
    for (int kt = 0; kt < 2; ++kt) {
      v16h a = load_a_frag(t1 + rbase * 72, 72, kt);
      const _Float16* wk = wt + (4 + kt * 4) * 512;
      a0 = wmma_f16(a, load_b_frag(wk + 0 * 512), a0);
      a1 = wmma_f16(a, load_b_frag(wk + 1 * 512), a1);
      a2 = wmma_f16(a, load_b_frag(wk + 2 * 512), a2);
      a3 = wmma_f16(a, load_b_frag(wk + 3 * 512), a3);
    }
    store_tile(t2, 72, rbase, 0, a0, b2, true);
    store_tile(t2, 72, rbase, 16, a1, b2, true);
    store_tile(t2, 72, rbase, 32, a2, b2, true);
    store_tile(t2, 72, rbase, 48, a3, b2, true);
  }

  float partial = 0.f;
  {  // L3: K=64(2 kt), 1 n-tile, 3 valid cols
    v8f acc = {};
    acc = wmma_f16(load_a_frag(t2 + rbase * 72, 72, 0), load_b_frag(wt + 12 * 512), acc);
    acc = wmma_f16(load_a_frag(t2 + rbase * 72, 72, 1), load_b_frag(wt + 13 * 512), acc);
    const int n = lane & 15, hi = lane >> 4;
    if (n < 3) {
      const float bb = b3[n];
      #pragma unroll
      for (int g = 0; g < 8; ++g) {
        const int row = R0 + rbase + g + 8 * hi;
        const float s = 1.f / (1.f + __expf(-(acc[g] + bb)));
        const float dif = hFeat[(size_t)row * 6 + n];
        dOut[1 + (size_t)3 * N + (size_t)row * 3 + n] = s;                  // spec
        dOut[1 + (size_t)row * 3 + n] = fminf(fmaxf(s + dif, 0.f), 1.f);    // rgbs
        partial += fminf(fmaxf(s + dif - 1.f, 0.f), 1.f);                   // sum1
      }
    }
  }
  red[tid] = partial;
  __syncthreads();
  #pragma unroll
  for (int s = 128; s > 0; s >>= 1) {
    if (tid < s) red[tid] += red[tid + s];
    __syncthreads();
  }
  if (tid == 0) parts[blockIdx.x] = red[0];
}

__global__ __launch_bounds__(256) void reduce_kernel(const float* __restrict__ parts,
                                                     int nParts,
                                                     float* __restrict__ out0,
                                                     float invCnt) {
  __shared__ float red[256];
  const int tid = threadIdx.x;
  float s = 0.f;
  for (int i = tid; i < nParts; i += 256) s += parts[i];
  red[tid] = s;
  __syncthreads();
  #pragma unroll
  for (int k = 128; k > 0; k >>= 1) {
    if (tid < k) red[tid] += red[tid + k];
    __syncthreads();
  }
  if (tid == 0) out0[0] = red[0] * invCnt;
}

// ---------------------------------------------------------------------------
// Host launcher
// ---------------------------------------------------------------------------
extern "C" void kernel_launch(void* const* d_in, const int* in_sizes, int n_in,
                              void* d_out, int out_size, void* d_ws, size_t ws_size,
                              hipStream_t stream) {
  const int N = in_sizes[0] / 3;  // 262144

  // d_in leaf order: x, normal, d, grid_table, then (W,b)* per MLP
  const float* x    = (const float*)d_in[0];
  const float* nrm  = (const float*)d_in[1];
  const float* dv   = (const float*)d_in[2];
  const float* grid = (const float*)d_in[3];
  const float* hW1 = (const float*)d_in[4];   const float* hb1 = (const float*)d_in[5];
  const float* hW2 = (const float*)d_in[6];   const float* hb2 = (const float*)d_in[7];
  const float* eW[5], *eb[5], *fW[5], *fb[5];
  for (int l = 0; l < 5; ++l) {
    eW[l] = (const float*)d_in[8 + 2 * l];   eb[l] = (const float*)d_in[9 + 2 * l];
    fW[l] = (const float*)d_in[18 + 2 * l];  fb[l] = (const float*)d_in[19 + 2 * l];
  }
  const float* rW1 = (const float*)d_in[28];  const float* rb1 = (const float*)d_in[29];
  const float* rW2 = (const float*)d_in[30];  const float* rb2 = (const float*)d_in[31];
  const float* rW3 = (const float*)d_in[32];  const float* rb3 = (const float*)d_in[33];
  float* out = (float*)d_out;

  // workspace carve-up
  char* ws = (char*)d_ws;
  size_t off = 0;
  auto take = [&](size_t bytes) {
    size_t o = off;
    off = (off + bytes + 255) & ~(size_t)255;
    return o;
  };
  _Float16* pe   = (_Float16*)(ws + take((size_t)N * 64 * 2));
  _Float16* feat = (_Float16*)(ws + take((size_t)N * 32 * 2));
  float*    ndir = (float*)(ws + take((size_t)N * 4));
  int*      mask = (int*)(ws + take((size_t)N * 4));
  float*    hbuf = (float*)(ws + take((size_t)N * 6 * 4));
  float*    envb = (float*)(ws + take((size_t)N * 3 * 4));
  float*    part = (float*)(ws + take((size_t)(N / 128) * 4));
  _Float16* wt   = (_Float16*)(ws + take((size_t)868 * 512 * 2));
  (void)ws_size; (void)n_in; (void)out_size;

  // ---- weight pre-tiling (block offsets in units of 512-half blocks) ----
  auto tileW = [&](const float* W, int K, int Nd, size_t blk, int kt, int nt) {
    tile_weights_kernel<<<dim3(kt * nt), dim3(512), 0, stream>>>(
        W, K, Nd, wt + blk * 512, nt);
  };
  tileW(hW1,  32,  64,   0, 1,  4);
  tileW(hW2,  64,   6,   4, 2,  1);
  tileW(eW[0], 36, 256,   6, 2, 16);
  tileW(eW[1], 256, 256, 38, 8, 16);
  tileW(eW[2], 256, 256, 166, 8, 16);
  tileW(eW[3], 256, 256, 294, 8, 16);
  tileW(eW[4], 256, 3,   422, 8,  1);
  tileW(fW[0], 36, 256,  430, 2, 16);
  tileW(fW[1], 256, 256, 462, 8, 16);
  tileW(fW[2], 256, 256, 590, 8, 16);
  tileW(fW[3], 256, 256, 718, 8, 16);
  tileW(fW[4], 256, 3,   846, 8,  1);
  tileW(rW1,   7,  64,   854, 1,  4);
  tileW(rW2,  64,  64,   858, 2,  4);
  tileW(rW3,  64,   3,   866, 2,  1);

  // ---- encode ----
  encode_kernel<<<dim3(N / 256), dim3(256), 0, stream>>>(
      x, nrm, dv, grid, pe, feat, ndir, mask, N);

  // ---- hash MLP (writes h + diffuse) ----
  hash_mlp_kernel<<<dim3(N / 128), dim3(256), 0, stream>>>(
      feat, wt, hb1, hb2, hbuf, out, N);

  // ---- env MLPs, one launch per branch ----
  Bias5 be{eb[0], eb[1], eb[2], eb[3], eb[4]};
  Bias5 bf{fb[0], fb[1], fb[2], fb[3], fb[4]};
  env_mlp_kernel<<<dim3(N / 128), dim3(256), 0, stream>>>(
      pe, wt + (size_t)6 * 512, be, mask, envb, 1);
  env_mlp_kernel<<<dim3(N / 128), dim3(256), 0, stream>>>(
      pe, wt + (size_t)430 * 512, bf, mask, envb, 0);

  // ---- rgb MLP + combine ----
  rgb_mlp_kernel<<<dim3(N / 128), dim3(256), 0, stream>>>(
      envb, hbuf, ndir, wt + (size_t)854 * 512, rb1, rb2, rb3, out, part, N);

  // ---- final deterministic mean ----
  reduce_kernel<<<dim3(1), dim3(256), 0, stream>>>(
      part, N / 128, out, 1.f / (float)(3 * (size_t)N));
}
